// LightweightMambaCodec_89129161327128
// MI455X (gfx1250) — compile-verified, matrix operand order checked by reading, not run on hardware
//
#include <hip/hip_runtime.h>
#include <hip/hip_bf16.h>
#include <cstddef>

// Model dims (compile-time constants from the reference)
#define B_  16
#define L_  512
#define DIN_ 64
#define DM_ 256
#define DI_ 512
#define NL_ 2
#define LAT_ 128
#define N_  16
#define K_  4
#define R_  16
#define BL_ (B_ * L_)   // 8192 rows for all big GEMMs
#define ROWW_ (R_ + 2 * N_)   // 48 floats per scan row (dt-in | B | C)

typedef __attribute__((ext_vector_type(16))) _Float16 v16h;
typedef __attribute__((ext_vector_type(8)))  float    v8f;
typedef int v4i_ __attribute__((vector_size(16)));   // matches builtin proto

// Async global->LDS staging (gfx1250, tracked by ASYNCcnt). Guarded so the
// file still compiles (with a synchronous fallback) if the toolchain lacks
// the builtins.
#if defined(__has_builtin)
#if __has_builtin(__builtin_amdgcn_global_load_async_to_lds_b128) && \
    __has_builtin(__builtin_amdgcn_s_wait_asynccnt)
#define HAVE_ASYNC_LDS 1
#endif
#endif
#ifndef HAVE_ASYNC_LDS
#define HAVE_ASYNC_LDS 0
#endif

#if HAVE_ASYNC_LDS
#define ASYNC_WAIT() __builtin_amdgcn_s_wait_asynccnt(0)
#else
#define ASYNC_WAIT()
#endif

// ---------------------------------------------------------------------------
// WMMA GEMM: C[M,N] = A[M,K] (f32) * W[N,K]^T (f32) (+ bias[N])
//
// One wave computes a 16 x (16*NT) strip: the A fragment is loaded/converted
// once per k-step and reused by NT back-to-back v_wmma_f32_16x16x32_f16 ops
// (NT-x better A-operand intensity; WMMAs co-execute on the XDL pipe while
// the next k-step's loads/converts issue). KDIM is compile-time so the
// k-loop is fully resolvable (KDIM % 32 == 0 for every GEMM in this model).
//
// CDNA5 fragment layouts (ISA 7.12.2):
//   A (16x32 f16): lane m = lane&15; lanes 0-15 hold K {0..7,16..23},
//                  lanes 16-31 hold K {8..15,24..31} (elements 0..7 / 8..15)
//   B (32x16 f16): lane n = lane&15; same K pattern (W is (N,K) row-major,
//                  so each 8-element run is contiguous -> b128 loads)
//   C/D (16x16 f32, 8 VGPRs): lane n = lane&15; elem r -> row r + 8*(lane>=16)
// ---------------------------------------------------------------------------
template <int KDIM, int NT>
__global__ void wmma_gemm_f32(const float* __restrict__ A,
                              const float* __restrict__ W,
                              const float* __restrict__ bias,
                              float* __restrict__ C,
                              int M, int N) {
    const int lane = threadIdx.x & 31;
    const int tm   = blockIdx.x * 16;
    const int tn   = blockIdx.y * (16 * NT);
    const int half = lane >> 4;          // 0: lanes 0-15, 1: lanes 16-31
    const int idx  = lane & 15;
    const int kc   = half * 8;           // K sub-offset per lane group

    const float* __restrict__ arow = A + (size_t)(tm + idx) * KDIM + kc;
    const float* wrow[NT];
#pragma unroll
    for (int t = 0; t < NT; ++t)
        wrow[t] = W + (size_t)(tn + 16 * t + idx) * KDIM + kc;

    v8f acc[NT];
#pragma unroll
    for (int t = 0; t < NT; ++t) acc[t] = v8f{};

    for (int k0 = 0; k0 < KDIM; k0 += 32) {
        v16h a;
        const float* ap = arow + k0;
#pragma unroll
        for (int e = 0; e < 8; ++e) {
            a[e]     = (_Float16)ap[e];
            a[e + 8] = (_Float16)ap[e + 16];
        }
#pragma unroll
        for (int t = 0; t < NT; ++t) {
            v16h b;
            const float* wp = wrow[t] + k0;
#pragma unroll
            for (int e = 0; e < 8; ++e) {
                b[e]     = (_Float16)wp[e];
                b[e + 8] = (_Float16)wp[e + 16];
            }
            acc[t] = __builtin_amdgcn_wmma_f32_16x16x32_f16(
                /*neg_a=*/false, a, /*neg_b=*/false, b,
                /*c_mod=*/(short)0, acc[t], /*reuse_a=*/false, /*reuse_b=*/false);
        }
    }

#pragma unroll
    for (int t = 0; t < NT; ++t) {
        const int cn = tn + 16 * t + idx;
        const float bv = bias ? bias[cn] : 0.0f;
#pragma unroll
        for (int r = 0; r < 8; ++r) {
            const int mr = tm + r + half * 8;
            C[(size_t)mr * N + cn] = acc[t][r] + bv;
        }
    }
}

// ---------------------------------------------------------------------------
// Fused depthwise causal conv (K=4) + SiLU on xc, and SiLU on the gate z.
// xz: (B,L,2*DI) from in-projection. Writes xc_act (B,L,DI) and sz (B,L,DI).
// ---------------------------------------------------------------------------
__global__ void conv_silu_kernel(const float* __restrict__ xz,
                                 const float* __restrict__ convw,  // (DI,1,K)
                                 const float* __restrict__ convb,  // (DI)
                                 float* __restrict__ xc_act,
                                 float* __restrict__ sz) {
    const int t = blockIdx.x * blockDim.x + threadIdx.x;
    if (t >= BL_ * DI_) return;
    const int d  = t % DI_;
    const int bl = t / DI_;
    const int l  = bl % L_;
    const int b  = bl / L_;

    float acc = convb[d];
#pragma unroll
    for (int j = 0; j < K_; ++j) {
        const int ls = l - (K_ - 1) + j;
        if (ls >= 0)
            acc += convw[d * K_ + j] *
                   xz[((size_t)b * L_ + ls) * (2 * DI_) + d];
    }
    xc_act[t] = acc / (1.0f + __expf(-acc));          // silu

    const float zv = xz[((size_t)b * L_ + l) * (2 * DI_) + DI_ + d];
    sz[t] = zv / (1.0f + __expf(-zv));                // silu(z)
}

// ---------------------------------------------------------------------------
// Fused dt-projection + softplus + selective scan + skip (Dp) + gate.
// One block per batch element, 512 threads = one channel d each (16 waves).
//
// This loop is the serial latency chain of the model. Per step l the
// 48-float row (dt-input[16] | B[16] | C[16]) is double-buffered in LDS:
// right after the per-step barrier, 12 lanes issue
// global_load_async_to_lds_b128 for row l+1 (tracked by ASYNCcnt), so the
// global->LDS latency overlaps the current step's 16 __expf's + dot product.
// Completion is enforced with s_wait_asynccnt 0 before the single per-step
// barrier. All 512 threads broadcast-read the row from LDS (same-address
// ds_load = conflict-free). State h[N=16] lives in VGPRs; next-step xc/sz
// rows are pulled in via global_prefetch.
// Writes g[b,l,d] = (y + u*Dp) * silu(z), the input to the out-projection.
// ---------------------------------------------------------------------------
__global__ void mamba_scan_kernel(const float* __restrict__ xdbl,  // (B,L,48)
                                  const float* __restrict__ xc,    // (B,L,DI)
                                  const float* __restrict__ sz,    // (B,L,DI)
                                  const float* __restrict__ Wdt,   // (DI,R)
                                  const float* __restrict__ bdt,   // (DI)
                                  const float* __restrict__ Alog,  // (DI,N)
                                  const float* __restrict__ Dp,    // (DI)
                                  float* __restrict__ g) {         // (B,L,DI)
    __shared__ __align__(16) float rows[2][ROWW_];   // 2 x 48 floats
    const int b = blockIdx.x;
    const int d = threadIdx.x;           // 0..511

    float wdt[R_], A[N_], h[N_];
#pragma unroll
    for (int i = 0; i < R_; ++i) wdt[i] = Wdt[d * R_ + i];
#pragma unroll
    for (int n = 0; n < N_; ++n) {
        A[n] = -__expf(Alog[d * N_ + n]);
        h[n] = 0.0f;
    }
    const float bd = bdt[d];
    const float dp = Dp[d];

    // Stage row `l` into LDS buffer `which`.
    auto stage = [&](int l, int which) {
#if HAVE_ASYNC_LDS
        if (d < (ROWW_ * 4) / 16) {   // 12 lanes x 16 bytes
            const float* gsrc = xdbl + ((size_t)b * L_ + l) * ROWW_ + d * 4;
            __builtin_amdgcn_global_load_async_to_lds_b128(
                (__attribute__((address_space(1))) v4i_*)(void*)gsrc,
                (__attribute__((address_space(3))) v4i_*)(void*)&rows[which][d * 4],
                0, 0);
        }
#else
        if (d < ROWW_)
            rows[which][d] = xdbl[((size_t)b * L_ + l) * ROWW_ + d];
#endif
    };

    int buf = 0;
    stage(0, 0);

    for (int l = 0; l < L_; ++l) {
        const size_t base = (size_t)b * L_ + l;
        ASYNC_WAIT();            // row `l` landed in rows[buf]
        __syncthreads();         // ... and is visible to all 16 waves

        if (l + 1 < L_) {
            stage(l + 1, buf ^ 1);   // overlap next row's fetch with compute
            __builtin_prefetch(&xc[(base + 1) * DI_ + d], 0, 0);
            __builtin_prefetch(&sz[(base + 1) * DI_ + d], 0, 0);
        }

        const float* rw = rows[buf];
        float dtr = bd;
#pragma unroll
        for (int i = 0; i < R_; ++i) dtr += rw[i] * wdt[i];
        const float dt = (dtr > 20.0f) ? dtr : __logf(1.0f + __expf(dtr));

        const float u  = xc[base * DI_ + d];
        const float du = dt * u;
        float y = 0.0f;
#pragma unroll
        for (int n = 0; n < N_; ++n) {
            h[n] = __expf(dt * A[n]) * h[n] + du * rw[R_ + n];
            y   += h[n] * rw[R_ + N_ + n];
        }
        y += u * dp;
        g[base * DI_ + d] = y * sz[base * DI_ + d];
        buf ^= 1;
        // No trailing barrier needed: the next iteration's wait+barrier
        // orders rows[buf^1] writes (issued above) against its readers.
    }
}

// ---------------------------------------------------------------------------
// pooled[b,c] = mean over l of h[b,l,c]
// ---------------------------------------------------------------------------
__global__ void mean_pool_kernel(const float* __restrict__ h,
                                 float* __restrict__ pooled) {
    const int t = blockIdx.x * blockDim.x + threadIdx.x;   // B_*DM_
    if (t >= B_ * DM_) return;
    const int c = t % DM_;
    const int b = t / DM_;
    float s = 0.0f;
    for (int l = 0; l < L_; ++l)
        s += h[((size_t)b * L_ + l) * DM_ + c];
    pooled[t] = s * (1.0f / (float)L_);
}

// ---------------------------------------------------------------------------
// Tiny GEMM for the bottleneck: out[b,j] = in[b,:]·W[j,:] + bias[j]
// Optionally mirrors the result to a second destination (latent -> d_out).
// ---------------------------------------------------------------------------
__global__ void small_gemm_kernel(const float* __restrict__ in,
                                  const float* __restrict__ W,
                                  const float* __restrict__ bias,
                                  float* __restrict__ out,
                                  float* __restrict__ out2,
                                  int Kdim, int Jout) {
    const int t = blockIdx.x * blockDim.x + threadIdx.x;
    if (t >= B_ * Jout) return;
    const int j = t % Jout;
    const int b = t / Jout;
    float s = bias[j];
    for (int k = 0; k < Kdim; ++k)
        s += in[b * Kdim + k] * W[j * Kdim + k];
    out[t] = s;
    if (out2) out2[t] = s;
}

// s[b,l,c] = srow[b,c]
__global__ void broadcast_rows_kernel(const float* __restrict__ srow,
                                      float* __restrict__ s) {
    const int t = blockIdx.x * blockDim.x + threadIdx.x;   // B_*L_*DM_
    if (t >= BL_ * DM_) return;
    const int c  = t % DM_;
    const int bl = t / DM_;
    const int b  = bl / L_;
    s[t] = srow[b * DM_ + c];
}

// ---------------------------------------------------------------------------
// Host-side orchestration
// ---------------------------------------------------------------------------
struct MambaW {
    const float *Win, *convw, *convb, *Wx, *Wdt, *bdt, *Alog, *Dp, *Wout;
};

static void run_mamba_layer(const float* in, float* out, const MambaW& w,
                            float* xz, float* xc, float* szb, float* xdbl,
                            float* g, hipStream_t stream) {
    // 1) in-projection: xz = in @ Win.T   (8192 x 256 -> 1024), 4 N-tiles/wave
    wmma_gemm_f32<DM_, 4><<<dim3(BL_ / 16, (2 * DI_) / 64), 32, 0, stream>>>(
        in, w.Win, nullptr, xz, BL_, 2 * DI_);
    // 2) causal depthwise conv + SiLU (+ gate SiLU)
    conv_silu_kernel<<<(BL_ * DI_ + 255) / 256, 256, 0, stream>>>(
        xz, w.convw, w.convb, xc, szb);
    // 3) x-projection: xdbl = xc @ Wx.T   (8192 x 512 -> 48), 3 N-tiles/wave
    wmma_gemm_f32<DI_, 3><<<dim3(BL_ / 16, 1), 32, 0, stream>>>(
        xc, w.Wx, nullptr, xdbl, BL_, R_ + 2 * N_);
    // 4) fused dt/softplus/scan/skip/gate (async double-buffered rows)
    mamba_scan_kernel<<<B_, DI_, 0, stream>>>(
        xdbl, xc, szb, w.Wdt, w.bdt, w.Alog, w.Dp, g);
    // 5) out-projection: out = g @ Wout.T (8192 x 512 -> 256), 4 N-tiles/wave
    wmma_gemm_f32<DI_, 4><<<dim3(BL_ / 16, DM_ / 64), 32, 0, stream>>>(
        g, w.Wout, nullptr, out, BL_, DM_);
}

extern "C" void kernel_launch(void* const* d_in, const int* in_sizes, int n_in,
                              void* d_out, int out_size, void* d_ws, size_t ws_size,
                              hipStream_t stream) {
    // Inputs (setup_inputs order)
    const float* x      = (const float*)d_in[0];   // (B,L,DIN)
    const float* emb_W  = (const float*)d_in[1];   // (DM,DIN)
    const float* emb_b  = (const float*)d_in[2];
    const float* lat_W  = (const float*)d_in[3];   // (LAT,DM)
    const float* lat_b  = (const float*)d_in[4];
    const float* l2s_W  = (const float*)d_in[5];   // (DM,LAT)
    const float* l2s_b  = (const float*)d_in[6];
    const float* out_W  = (const float*)d_in[7];   // (DIN,DM)
    const float* out_b  = (const float*)d_in[8];

    const float* enc[9];
    const float* dec[9];
    for (int i = 0; i < 9; ++i) enc[i] = (const float*)d_in[9 + i];
    for (int i = 0; i < 9; ++i) dec[i] = (const float*)d_in[18 + i];

    // Per-layer strides within stacked (NL, ...) weights
    const size_t st_Win = (size_t)(2 * DI_) * DM_;
    const size_t st_cw  = (size_t)DI_ * K_;
    const size_t st_cb  = DI_;
    const size_t st_Wx  = (size_t)(R_ + 2 * N_) * DI_;
    const size_t st_Wdt = (size_t)DI_ * R_;
    const size_t st_bdt = DI_;
    const size_t st_Al  = (size_t)DI_ * N_;
    const size_t st_Dp  = DI_;
    const size_t st_Wo  = (size_t)DM_ * DI_;

    // Workspace carve-up (floats; all offsets keep 16B alignment)
    float* ws = (float*)d_ws;
    float* h0     = ws;                      // B*L*DM
    float* h1     = h0 + (size_t)BL_ * DM_;  // B*L*DM
    float* xz     = h1 + (size_t)BL_ * DM_;  // B*L*2*DI
    float* xc     = xz + (size_t)BL_ * 2 * DI_;
    float* szb    = xc + (size_t)BL_ * DI_;
    float* xdbl   = szb + (size_t)BL_ * DI_; // B*L*48
    float* g      = xdbl + (size_t)BL_ * ROWW_;
    float* pooled = g + (size_t)BL_ * DI_;   // B*DM
    float* latent = pooled + (size_t)B_ * DM_;   // B*LAT
    float* srow   = latent + (size_t)B_ * LAT_;  // B*DM

    float* recon_out  = (float*)d_out;                     // B*L*DIN
    float* latent_out = recon_out + (size_t)BL_ * DIN_;    // B*LAT

    // ---- Embedding: h0 = x @ emb_W.T + emb_b  (K=64, 4 N-tiles/wave) ----
    wmma_gemm_f32<DIN_, 4><<<dim3(BL_ / 16, DM_ / 64), 32, 0, stream>>>(
        x, emb_W, emb_b, h0, BL_, DM_);

    // ---- Encoder layers ----
    float* cur = h0;
    float* nxt = h1;
    for (int i = 0; i < NL_; ++i) {
        MambaW w = { enc[0] + i * st_Win, enc[1] + i * st_cw, enc[2] + i * st_cb,
                     enc[3] + i * st_Wx,  enc[4] + i * st_Wdt, enc[5] + i * st_bdt,
                     enc[6] + i * st_Al,  enc[7] + i * st_Dp,  enc[8] + i * st_Wo };
        run_mamba_layer(cur, nxt, w, xz, xc, szb, xdbl, g, stream);
        float* t = cur; cur = nxt; nxt = t;
    }
    // cur now holds encoder output (h0 for NL=2)

    // ---- Bottleneck ----
    mean_pool_kernel<<<(B_ * DM_ + 255) / 256, 256, 0, stream>>>(cur, pooled);
    small_gemm_kernel<<<(B_ * LAT_ + 63) / 64, 64, 0, stream>>>(
        pooled, lat_W, lat_b, latent, latent_out, DM_, LAT_);
    small_gemm_kernel<<<(B_ * DM_ + 63) / 64, 64, 0, stream>>>(
        latent, l2s_W, l2s_b, srow, nullptr, LAT_, DM_);
    broadcast_rows_kernel<<<(BL_ * DM_ + 255) / 256, 256, 0, stream>>>(srow, cur);

    // ---- Decoder layers ----
    for (int i = 0; i < NL_; ++i) {
        MambaW w = { dec[0] + i * st_Win, dec[1] + i * st_cw, dec[2] + i * st_cb,
                     dec[3] + i * st_Wx,  dec[4] + i * st_Wdt, dec[5] + i * st_bdt,
                     dec[6] + i * st_Al,  dec[7] + i * st_Dp,  dec[8] + i * st_Wo };
        run_mamba_layer(cur, nxt, w, xz, xc, szb, xdbl, g, stream);
        float* t = cur; cur = nxt; nxt = t;
    }

    // ---- Reconstruction: recon = s @ out_W.T + out_b (K=256, N=64) ----
    wmma_gemm_f32<DM_, 4><<<dim3(BL_ / 16, 1), 32, 0, stream>>>(
        cur, out_W, out_b, recon_out, BL_, DIN_);
}
